// CausalSelfAttention_65627100283381
// MI455X (gfx1250) — matrix operand order repaired and checked by reading, not compiled, not run on hardware
//
#include <hip/hip_runtime.h>
#include <hip/hip_bf16.h>

// Problem constants (B=2, T=2048, C=2048, H=16, D=128)
#define BB 2
#define TT 2048
#define CC 2048
#define HH 16
#define DD 128

typedef __bf16 v8bf  __attribute__((ext_vector_type(8)));
typedef __bf16 v16bf __attribute__((ext_vector_type(16)));
typedef float  v8f   __attribute__((ext_vector_type(8)));
typedef unsigned int v4u __attribute__((ext_vector_type(4)));
typedef int v4i __attribute__((ext_vector_type(4)));
typedef int v8i __attribute__((ext_vector_type(8)));

#if __has_builtin(__builtin_amdgcn_tensor_load_to_lds)
#define HAVE_TDM 1
#else
#define HAVE_TDM 0
#endif

// LDS B-tile: 64 rows x 128 k-elements bf16, padded 272B row stride (136 el)
// to spread rows across banks (TDM pad: 4 DWORDs after every 64 DWORDs).
#define BROW_EL 136
#define BCHUNK 128

__device__ __forceinline__ v16bf cat16(v8bf lo, v8bf hi) {
  return __builtin_shufflevector(lo, hi, 0,1,2,3,4,5,6,7,8,9,10,11,12,13,14,15);
}

#if HAVE_TDM
// Issue one TDM 2D tile load: 64 rows x 128 bf16 from row-major [N=2048,K=2048]
// weight matrix into LDS at lds_byte_addr, with per-row padding (272B stride).
// Descriptor packing per CDNA5 ISA.
__device__ __forceinline__ void tdm_load_tile(const __bf16* gptr, unsigned lds_byte_addr)
{
  unsigned long long ga = (unsigned long long)gptr;
  v4u g0;
  g0.x = 1u;                                          // count=1 (valid), user mode
  g0.y = lds_byte_addr;                               // [63:32] lds_addr
  g0.z = (unsigned)(ga & 0xFFFFFFFFull);              // [95:64] global_addr lo
  g0.w = (unsigned)((ga >> 32) & 0x1FFFFFFull)        // [120:96] global_addr hi
         | (2u << 30);                                // [127:126] type=2 (image)
  // dword0: data_size=1 (2B) <<16 | pad_en<<20 | pad_interval=5(64DW)<<22 | pad_amount=3(4DW)<<25
  v8i g1;
  g1[0] = (int)((1u << 16) | (1u << 20) | (5u << 22) | (3u << 25));
  g1[1] = (int)(2048u << 16);   // tensor_dim0 = 2048 (bits 48..79, low16 here)
  g1[2] = (int)(2048u << 16);   // tensor_dim0 hi=0 | tensor_dim1 = 2048 (low16)
  g1[3] = (int)(128u << 16);    // tensor_dim1 hi=0 | tile_dim0 = 128
  g1[4] = 64;                   // tile_dim1 = 64, tile_dim2 = 0
  g1[5] = 2048;                 // tensor_dim0_stride lo32 = 2048
  g1[6] = 0;                    // stride hi | tensor_dim1_stride lo (unused, 2D)
  g1[7] = 0;
  v4i z4 = {0, 0, 0, 0};
#if defined(__clang_major__) && (__clang_major__ >= 23)
  v8i z8 = {0, 0, 0, 0, 0, 0, 0, 0};
  __builtin_amdgcn_tensor_load_to_lds(g0, g1, z4, z4, z8, 0);
#else
  __builtin_amdgcn_tensor_load_to_lds(g0, g1, z4, z4, 0);
#endif
}
#endif

// ---------------------------------------------------------------------------
// Kernel 1: token-shift mix  xq/xk/xv = x + (shift(x)-x)*mix, converted to bf16
// ---------------------------------------------------------------------------
__global__ __launch_bounds__(256)
void prep_act_kernel(const float* __restrict__ x,
                     const float* __restrict__ mq,
                     const float* __restrict__ mk,
                     const float* __restrict__ mv,
                     __bf16* __restrict__ xq,
                     __bf16* __restrict__ xk,
                     __bf16* __restrict__ xv)
{
  size_t i = (size_t)blockIdx.x * blockDim.x + threadIdx.x;
  if (i >= (size_t)BB * TT * CC) return;
  int c = (int)(i & (CC - 1));
  int t = (int)((i >> 11) & (TT - 1));
  float xi = x[i];
  float dxp = (t > 0 ? x[i - CC] : 0.0f) - xi;
  xq[i] = (__bf16)(xi + dxp * mq[c]);
  xk[i] = (__bf16)(xi + dxp * mk[c]);
  xv[i] = (__bf16)(xi + dxp * mv[c]);
}

// ---------------------------------------------------------------------------
// Kernel 2: weight conversion f32 -> bf16 (all four weights in one pass)
// ---------------------------------------------------------------------------
__global__ __launch_bounds__(256)
void prep_w_kernel(const float* __restrict__ Wq, const float* __restrict__ Wk,
                   const float* __restrict__ Wv, const float* __restrict__ Wp,
                   __bf16* __restrict__ wq, __bf16* __restrict__ wk,
                   __bf16* __restrict__ wv, __bf16* __restrict__ wp)
{
  size_t i = (size_t)blockIdx.x * blockDim.x + threadIdx.x;
  if (i >= (size_t)CC * CC) return;
  wq[i] = (__bf16)Wq[i];
  wk[i] = (__bf16)Wk[i];
  wv[i] = (__bf16)Wv[i];
  wp[i] = (__bf16)Wp[i];
}

// ---------------------------------------------------------------------------
// Kernel 3: WMMA GEMM  Out[M,N] = A[M,K] @ Bw[N,K]^T   (bf16 in, f32 acc)
// Block = 256 threads = 8 waves; block tile 128(M) x 64(N); wave tile 16x64.
// The 64-row weight slab is staged into LDS by the Tensor Data Mover,
// double-buffered in 128-wide K chunks (wave 0 issues, s_wait_tensorcnt
// pipelines, barrier publishes).  B-fragments then come from LDS
// (ds_load_b128, conflict-free via TDM row padding); A-fragments stream
// from global/L2 (unique per wave).
// mode 0: store bf16 row-major (q,k)
// mode 1: v-mix with lamb*v1, store transposed bf16 as [B,H,D,T]
// mode 2: residual + acc, store f32 (final projection)
// ---------------------------------------------------------------------------
__global__ __launch_bounds__(256)
void gemm_bf16_kernel(const __bf16* __restrict__ A,
                      const __bf16* __restrict__ Bw,
                      int Kdim, int Ndim, int mode,
                      __bf16* __restrict__ outbf,
                      __bf16* __restrict__ vTout,
                      const float* __restrict__ v1,
                      const float* __restrict__ lambp,
                      const float* __restrict__ residual,
                      float* __restrict__ outf)
{
  __shared__ __bf16 bsm[2][64 * BROW_EL];   // 2 x 17408 B double buffer

  const int lane   = threadIdx.x & 31;
  const int wv     = threadIdx.x >> 5;
  const int halfid = lane >> 4;
  const int lcol   = lane & 15;
  const int m0 = blockIdx.y * 128 + wv * 16;
  const int n0 = blockIdx.x * 64;
  const int nch = Kdim / BCHUNK;

  v8f acc[4] = {};
  // A fragment base: lane = M row (lcol); 16 contiguous K at +16*halfid
  const __bf16* arow = A + (size_t)(m0 + lcol) * Kdim + halfid * 16;

#if HAVE_TDM
  if (wv == 0) {
    tdm_load_tile(Bw + (size_t)n0 * Kdim,
                  (unsigned)(unsigned long long)(const void*)&bsm[0][0]);
  }
#endif

  for (int kc = 0; kc < nch; ++kc) {
#if HAVE_TDM
    if (wv == 0) {
      if (kc + 1 < nch) {
        tdm_load_tile(Bw + (size_t)n0 * Kdim + (size_t)(kc + 1) * BCHUNK,
                      (unsigned)(unsigned long long)(const void*)&bsm[(kc + 1) & 1][0]);
        __builtin_amdgcn_s_wait_tensorcnt(1);   // chunk kc landed (in-order)
      } else {
        __builtin_amdgcn_s_wait_tensorcnt(0);
      }
    }
    __syncthreads();                            // publish chunk kc to all waves
    const __bf16* bbase = &bsm[kc & 1][0];
#else
    // Fallback staging: all 256 threads copy the 64x128 chunk (single buffer).
    {
      const __bf16* src = Bw + (size_t)n0 * Kdim + (size_t)kc * BCHUNK;
      __syncthreads();
#pragma unroll
      for (int i = 0; i < 4; ++i) {
        int idx = threadIdx.x + i * 256;        // 1024 chunks of 8 elements
        int r = idx >> 4, cc8 = (idx & 15) * 8;
        *(v8bf*)&bsm[0][r * BROW_EL + cc8] = *(const v8bf*)(src + (size_t)r * Kdim + cc8);
      }
      __syncthreads();
    }
    const __bf16* bbase = &bsm[0][0];
#endif

#pragma unroll
    for (int k2 = 0; k2 < 4; ++k2) {            // 4 WMMA K-steps per chunk
      const int kk = kc * BCHUNK + k2 * 32;
      __builtin_prefetch(arow + kk + 256, 0, 1);
      v16bf afrag = cat16(*(const v8bf*)(arow + kk),
                          *(const v8bf*)(arow + kk + 8));
#pragma unroll
      for (int j = 0; j < 4; ++j) {
        // B fragment from LDS: lane = N column; 16 contiguous K at k2*32+16*halfid
        const __bf16* bp = bbase + (j * 16 + lcol) * BROW_EL + k2 * 32 + halfid * 16;
        v16bf bfrag = cat16(*(const v8bf*)bp, *(const v8bf*)(bp + 8));
        acc[j] = __builtin_amdgcn_wmma_f32_16x16x32_bf16(
            false, afrag, false, bfrag, (short)0, acc[j], false, false);
      }
    }
    __syncthreads();                            // reads done before buffer reuse
  }

  const float lamb = (mode == 1) ? lambp[0] : 0.0f;
#pragma unroll
  for (int j = 0; j < 4; ++j) {
#pragma unroll
    for (int r = 0; r < 8; ++r) {
      int row = m0 + r + 8 * halfid;        // C/D layout: M = vgpr + 8*(lane/16)
      int n   = n0 + j * 16 + lcol;         // N = lane%16
      float v = acc[j][r];
      if (mode == 0) {
        outbf[(size_t)row * Ndim + n] = (__bf16)v;
      } else if (mode == 1) {
        float f = (1.0f - lamb) * v + lamb * v1[(size_t)row * Ndim + n];
        int b = row >> 11;                  // row / T
        int t = row & (TT - 1);
        int h = n >> 7;                     // n / D
        int d = n & (DD - 1);
        vTout[(((size_t)(b * HH + h)) * DD + d) * TT + t] = (__bf16)f;
      } else {
        size_t idx = (size_t)row * Ndim + n;
        outf[idx] = residual[idx] + v;
      }
    }
  }
}

// ---------------------------------------------------------------------------
// Kernel 4: fused RMS-norm + RoPE on q and k (bf16 in-place).
// One wave per (b,t,h) row of 128; lane handles d = {lane, lane+32} paired
// with d+64 (reference rotates concatenated halves).
// ---------------------------------------------------------------------------
__global__ __launch_bounds__(32)
void qk_norm_rope_kernel(__bf16* __restrict__ qb, __bf16* __restrict__ kb)
{
  __bf16* buf = blockIdx.y ? kb : qb;
  const int lane = threadIdx.x;
  const int idx  = blockIdx.x;             // b*T*H + t*H + h
  const int h = idx & (HH - 1);
  const int t = (idx >> 4) & (TT - 1);
  const int b = idx >> 15;
  __bf16* p = buf + ((size_t)(b * TT + t)) * CC + h * DD;

  float x1a = (float)p[lane];
  float x1b = (float)p[lane + 32];
  float x2a = (float)p[lane + 64];
  float x2b = (float)p[lane + 96];

  float ss = x1a * x1a + x1b * x1b + x2a * x2a + x2b * x2b;
  ss += __shfl_xor(ss, 1);
  ss += __shfl_xor(ss, 2);
  ss += __shfl_xor(ss, 4);
  ss += __shfl_xor(ss, 8);
  ss += __shfl_xor(ss, 16);
  float rn = rsqrtf(ss * (1.0f / 128.0f) + 1.1920929e-7f);  // f32 eps
  x1a *= rn; x1b *= rn; x2a *= rn; x2b *= rn;

  const float kln = 9.210340371976184f / 64.0f;  // ln(10000)/64
  float tf = (float)t;
  float ang0 = tf * __expf(-kln * (float)lane);
  float ang1 = tf * __expf(-kln * (float)(lane + 32));
  float s0, c0, s1, c1;
  __sincosf(ang0, &s0, &c0);
  __sincosf(ang1, &s1, &c1);

  p[lane]      = (__bf16)( x1a * c0 + x2a * s0);
  p[lane + 64] = (__bf16)(-x1a * s0 + x2a * c0);
  p[lane + 32] = (__bf16)( x1b * c1 + x2b * s1);
  p[lane + 96] = (__bf16)(-x1b * s1 + x2b * c1);
}

// ---------------------------------------------------------------------------
// Kernel 5: flash attention, one wave per 16-row Q tile, K-step = 32
// (= WMMA K for bf16).  Scores via 8 WMMAs/step, online softmax with
// half-lane row stats, P converted C-layout -> A-layout through 1KB LDS,
// PV via 8 WMMAs/step against transposed V [B,H,D,T].
// ---------------------------------------------------------------------------
__global__ __launch_bounds__(32)
void flash_attn_kernel(const __bf16* __restrict__ qb,
                       const __bf16* __restrict__ kb,
                       const __bf16* __restrict__ vT,
                       __bf16* __restrict__ ao)
{
  __shared__ __bf16 ptile[16 * 32];  // P tile staging (row-major 16 x 32)

  const int lane   = threadIdx.x & 31;
  const int halfid = lane >> 4;
  const int lcol   = lane & 15;
  const int qt = blockIdx.x;                 // q tile index
  const int bh = blockIdx.y;                 // b*H + h
  const int bb = bh >> 4;
  const int hh = bh & (HH - 1);
  const int qrow0 = qt * 16;

  const __bf16* qbase = qb + ((size_t)bb * TT) * CC + hh * DD;
  const __bf16* kbase = kb + ((size_t)bb * TT) * CC + hh * DD;
  const __bf16* vbase = vT + (size_t)bh * DD * TT;

  // Q A-fragments for the four 32-wide D chunks
  v16bf aq[4];
#pragma unroll
  for (int c = 0; c < 4; ++c) {
    const __bf16* qp = qbase + (size_t)(qrow0 + lcol) * CC + c * 32 + halfid * 16;
    aq[c] = cat16(*(const v8bf*)qp, *(const v8bf*)(qp + 8));
  }

  v8f accO[8] = {};
  float m8[8], l8[8];
#pragma unroll
  for (int r = 0; r < 8; ++r) { m8[r] = -1e30f; l8[r] = 0.0f; }

  const float scale = 0.08838834764831845f;  // 1/sqrt(128)
  const int nsteps = (qrow0 + 15) / 32 + 1;

  for (int ks = 0; ks < nsteps; ++ks) {
    const int t0 = ks * 32;
    v8f s0 = {}, s1 = {};
#pragma unroll
    for (int c = 0; c < 4; ++c) {
      const __bf16* kp0 = kbase + (size_t)(t0 + lcol) * CC + c * 32 + halfid * 16;
      const __bf16* kp1 = kbase + (size_t)(t0 + 16 + lcol) * CC + c * 32 + halfid * 16;
      v16bf kf0 = cat16(*(const v8bf*)kp0, *(const v8bf*)(kp0 + 8));
      v16bf kf1 = cat16(*(const v8bf*)kp1, *(const v8bf*)(kp1 + 8));
      s0 = __builtin_amdgcn_wmma_f32_16x16x32_bf16(false, aq[c], false, kf0, (short)0, s0, false, false);
      s1 = __builtin_amdgcn_wmma_f32_16x16x32_bf16(false, aq[c], false, kf1, (short)0, s1, false, false);
    }

    // online softmax (rows live in 16-lane halves; shfl masks 1..8 stay in-half)
    float p0a[8], p1a[8];
#pragma unroll
    for (int r = 0; r < 8; ++r) {
      int rowg = qrow0 + r + 8 * halfid;
      float a = s0[r] * scale;
      float bm = s1[r] * scale;
      if (t0 + lcol > rowg)      a  = -1e30f;   // causal mask
      if (t0 + 16 + lcol > rowg) bm = -1e30f;
      float mx = fmaxf(a, bm);
      mx = fmaxf(mx, __shfl_xor(mx, 1));
      mx = fmaxf(mx, __shfl_xor(mx, 2));
      mx = fmaxf(mx, __shfl_xor(mx, 4));
      mx = fmaxf(mx, __shfl_xor(mx, 8));
      float mn = fmaxf(m8[r], mx);
      float co = __expf(m8[r] - mn);
      float p0 = __expf(a - mn);
      float p1 = __expf(bm - mn);
      float rs = p0 + p1;
      rs += __shfl_xor(rs, 1);
      rs += __shfl_xor(rs, 2);
      rs += __shfl_xor(rs, 4);
      rs += __shfl_xor(rs, 8);
      l8[r] = l8[r] * co + rs;
      m8[r] = mn;
#pragma unroll
      for (int c2 = 0; c2 < 8; ++c2) accO[c2][r] *= co;
      p0a[r] = p0;
      p1a[r] = p1;
    }

    __syncthreads();  // previous iteration's ds_loads done before overwrite
#pragma unroll
    for (int r = 0; r < 8; ++r) {
      int rowl = r + 8 * halfid;
      ptile[rowl * 32 + lcol]      = (__bf16)p0a[r];
      ptile[rowl * 32 + 16 + lcol] = (__bf16)p1a[r];
    }
    __syncthreads();  // stores visible before A-layout reload

    // reload P as A-fragment (lane = M row; 16 contiguous K at +16*halfid)
    const __bf16* pp = &ptile[lcol * 32 + halfid * 16];
    v16bf ap = cat16(*(const v8bf*)pp, *(const v8bf*)(pp + 8));

    // PV: B fragment from transposed V — contiguous t per fixed d
#pragma unroll
    for (int c2 = 0; c2 < 8; ++c2) {
      const __bf16* vp = vbase + (size_t)(c2 * 16 + lcol) * TT + t0 + halfid * 16;
      v16bf vf = cat16(*(const v8bf*)vp, *(const v8bf*)(vp + 8));
      accO[c2] = __builtin_amdgcn_wmma_f32_16x16x32_bf16(false, ap, false, vf, (short)0, accO[c2], false, false);
    }
  }

  // epilogue: normalize by row sum, store bf16 [B,T,C]
#pragma unroll
  for (int r = 0; r < 8; ++r) {
    float inv = 1.0f / l8[r];
    int row = qrow0 + r + 8 * halfid;
    __bf16* op = ao + ((size_t)(bb * TT + row)) * CC + hh * DD;
#pragma unroll
    for (int c2 = 0; c2 < 8; ++c2) {
      op[c2 * 16 + lcol] = (__bf16)(accO[c2][r] * inv);
    }
  }
}

// ---------------------------------------------------------------------------
// Launcher
// ---------------------------------------------------------------------------
extern "C" void kernel_launch(void* const* d_in, const int* in_sizes, int n_in,
                              void* d_out, int out_size, void* d_ws, size_t ws_size,
                              hipStream_t stream) {
  (void)in_sizes; (void)n_in; (void)out_size; (void)ws_size;
  const float* residual = (const float*)d_in[0];
  const float* x        = (const float*)d_in[1];
  const float* v1       = (const float*)d_in[2];
  // d_in[3]=x0, d_in[4]=dx0, d_in[5]=token_ids : unused by reference math
  const float* x_q      = (const float*)d_in[6];
  const float* x_k      = (const float*)d_in[7];
  const float* x_v      = (const float*)d_in[8];
  const float* Wq       = (const float*)d_in[9];
  const float* Wk       = (const float*)d_in[10];
  const float* Wv       = (const float*)d_in[11];
  const float* Wp       = (const float*)d_in[12];
  const float* lamb     = (const float*)d_in[13];
  float* outf           = (float*)d_out;

  const size_t SACT = (size_t)BB * TT * CC;  // activation elements
  const size_t SW   = (size_t)CC * CC;       // weight elements

  __bf16* xq   = (__bf16*)d_ws;
  __bf16* xk   = xq + SACT;
  __bf16* xv   = xk + SACT;
  __bf16* wqb  = xv + SACT;
  __bf16* wkb  = wqb + SW;
  __bf16* wvb  = wkb + SW;
  __bf16* wpb  = wvb + SW;
  __bf16* qbuf = wpb + SW;
  __bf16* kbuf = qbuf + SACT;
  __bf16* vT   = kbuf + SACT;
  __bf16* ao   = vT + SACT;

  prep_act_kernel<<<(unsigned)((SACT + 255) / 256), 256, 0, stream>>>(
      x, x_q, x_k, x_v, xq, xk, xv);
  prep_w_kernel<<<(unsigned)((SW + 255) / 256), 256, 0, stream>>>(
      Wq, Wk, Wv, Wp, wqb, wkb, wvb, wpb);

  dim3 gg(CC / 64, (BB * TT) / 128);
  gemm_bf16_kernel<<<gg, 256, 0, stream>>>(xq, wqb, CC, CC, 0,
      qbuf, nullptr, nullptr, nullptr, nullptr, nullptr);
  gemm_bf16_kernel<<<gg, 256, 0, stream>>>(xk, wkb, CC, CC, 0,
      kbuf, nullptr, nullptr, nullptr, nullptr, nullptr);
  gemm_bf16_kernel<<<gg, 256, 0, stream>>>(xv, wvb, CC, CC, 1,
      nullptr, vT, v1, lamb, nullptr, nullptr);

  qk_norm_rope_kernel<<<dim3(BB * TT * HH, 2), 32, 0, stream>>>(qbuf, kbuf);

  flash_attn_kernel<<<dim3(TT / 16, BB * HH), 32, 0, stream>>>(qbuf, kbuf, vT, ao);

  gemm_bf16_kernel<<<gg, 256, 0, stream>>>(ao, wpb, CC, CC, 2,
      nullptr, nullptr, nullptr, nullptr, residual, outf);
}
